// CrossAttention_59115929862230
// MI455X (gfx1250) — compile-verified, matrix operand order checked by reading, not run on hardware
//
#include <hip/hip_runtime.h>

typedef __attribute__((ext_vector_type(16))) __bf16       bf16x16;
typedef __attribute__((ext_vector_type(8)))  float        f32x8;
typedef __attribute__((ext_vector_type(4)))  unsigned int u32x4;
typedef __attribute__((ext_vector_type(8)))  int          i32x8;
typedef __attribute__((ext_vector_type(4)))  int          i32x4;

#define BB   4
#define NN   1024
#define IND  768
#define OUTD 768
#define HH   12
#define DD   64
#define OUTC 2304          // 3*OUTD
#define ATT_SCALE 0.125f   // 1/sqrt(64)

static constexpr size_t QKV_ELEMS = (size_t)BB * HH * NN * DD;     // per q/k/v part
static constexpr size_t ATT_ELEMS = (size_t)BB * HH * NN * NN;     // per stream
static constexpr size_t HEAD_STRIDE = (size_t)NN * DD;             // q/k/v per-head
static constexpr size_t ATTH_STRIDE = (size_t)NN * NN;             // attn per-head

// ---------------------------------------------------------------------------
// Kernel 1: QKV projection for both streams.
//   [4096,768] @ [768,2304] + bias, f32 -> bf16 q/k/v in [t][part][B,H,N,D].
// Block = 4 waves covering a 16x64 output tile. The 32x64 W chunk is staged
// cooperatively in LDS (vectorized f32 loads -> bf16), then each wave reads
// its 32x16 B operand from LDS; A operand is a contiguous 64B load per lane.
// ---------------------------------------------------------------------------
__global__ __launch_bounds__(128)
void qkv_kernel(const float* __restrict__ X, const float* __restrict__ Y,
                const float* __restrict__ Wx, const float* __restrict__ bx,
                const float* __restrict__ Wy, const float* __restrict__ by,
                __bf16* __restrict__ qkv)
{
    __shared__ __bf16 wtile[32][64];                 // 4 KB

    const int lane = threadIdx.x;
    const int wave = threadIdx.y;                    // 0..3 -> column sub-tile
    const int tid  = wave * 32 + lane;
    const int half = lane >> 4;
    const int lm   = lane & 15;
    const int r0   = blockIdx.x * 16;                // row tile in [B*N]
    const int jblk = blockIdx.y * 64;                // 64-column block
    const int t    = blockIdx.z;                     // 0:x 1:y

    const float* A    = t ? Y  : X;
    const float* W    = t ? Wy : Wx;
    const float* bias = t ? by : bx;

    // cooperative W loader: thread -> (row 0..31, 16-col group)
    const int wrow = tid >> 2;
    const int wcol = (tid & 3) * 16;

    const int j = jblk + wave * 16 + lm;
    f32x8 acc = {};

    for (int kk = 0; kk < IND; kk += 32) {
        // ---- stage W chunk [32 x 64] into LDS (bf16) ----
        const float* wp = W + (size_t)(kk + wrow) * OUTC + jblk + wcol;
        bf16x16 wv;
        #pragma unroll
        for (int e = 0; e < 16; ++e) wv[e] = (__bf16)wp[e];

        // ---- A tile (16x32 bf16): 16 contiguous K per lane ----
        const float* ap = A + (size_t)(r0 + lm) * IND + kk + 16 * half;
        bf16x16 a;
        #pragma unroll
        for (int e = 0; e < 16; ++e) a[e] = (__bf16)ap[e];

        __syncthreads();                              // WAR vs previous reads
        *(bf16x16*)&wtile[wrow][wcol] = wv;           // 32B aligned ds_store
        __syncthreads();                              // tile visible

        bf16x16 b;                                    // B operand from LDS
        #pragma unroll
        for (int e = 0; e < 16; ++e) b[e] = wtile[16 * half + e][wave * 16 + lm];

        acc = __builtin_amdgcn_wmma_f32_16x16x32_bf16(false, a, false, b,
                                                      (short)0, acc, false, false);
        if (kk + 32 < IND) __builtin_prefetch(ap + 32, 0, 3);
    }

    const float bv   = bias[j];
    const int   part = j / OUTD;          // 0=q 1=k 2=v (16-col tile never crosses)
    const int   jj   = j - part * OUTD;
    const int   h    = jj >> 6;
    const int   d    = jj & 63;

    // 16-row tile never crosses a batch boundary (1024 % 16 == 0):
    // hoist the (b,n) decomposition, store with constant row stride DD.
    const int   b_ = r0 >> 10;
    const int   nb = (r0 & 1023) + 8 * half;
    __bf16* dp = qkv + ((size_t)t * 3 + part) * QKV_ELEMS
               + (((size_t)(b_ * HH + h)) * NN + nb) * DD + d;
    #pragma unroll
    for (int i = 0; i < 8; ++i)           // D tile: M = i + 8*half, N = lm
        dp[(size_t)i * DD] = (__bf16)(acc[i] + bv);
}

// ---------------------------------------------------------------------------
// Kernel 2: fused scores + gamma cross-mix + softmax over the HEAD axis.
// One wave owns a 16x16 (n,m) tile for ALL 12 heads of BOTH streams:
// 48 bf16 WMMAs accumulate into 24 v8f regs; head-softmax is pure per-lane
// VALU across the accumulators. fp32 logits never touch memory.
// ---------------------------------------------------------------------------
__global__ __launch_bounds__(128)
void scores_softmax_kernel(const __bf16* __restrict__ qkv,
                           const float* __restrict__ g1p,
                           const float* __restrict__ g2p,
                           __bf16* __restrict__ attn)
{
    const int lane = threadIdx.x;
    const int half = lane >> 4;
    const int lm   = lane & 15;
    const int n0 = blockIdx.x * 16;
    const int m0 = (blockIdx.y * blockDim.y + threadIdx.y) * 16;
    const int b  = blockIdx.z;
    const float g1 = g1p[0];
    const float g2 = g2p[0];

    const __bf16* qx = qkv + 0 * QKV_ELEMS;
    const __bf16* kx = qkv + 1 * QKV_ELEMS;
    const __bf16* qy = qkv + 3 * QKV_ELEMS;
    const __bf16* ky = qkv + 4 * QKV_ELEMS;

    f32x8 ax[HH], ay[HH];
    #pragma unroll
    for (int h = 0; h < HH; ++h) { ax[h] = f32x8{}; ay[h] = f32x8{}; }

    // base offsets for head 0; per-head step is the constant HEAD_STRIDE
    const size_t qoff0 = ((size_t)(b * HH) * NN + n0 + lm) * DD + 16 * half;
    const size_t koff0 = ((size_t)(b * HH) * NN + m0 + lm) * DD + 16 * half;

    #pragma unroll
    for (int h = 0; h < HH; ++h) {
        const size_t qoff = qoff0 + (size_t)h * HEAD_STRIDE;
        const size_t koff = koff0 + (size_t)h * HEAD_STRIDE;
        #pragma unroll
        for (int c = 0; c < 2; ++c) {     // D=64 -> two K=32 chunks
            bf16x16 aqx = *(const bf16x16*)(qx + qoff + c * 32);
            bf16x16 bkx = *(const bf16x16*)(kx + koff + c * 32);
            ax[h] = __builtin_amdgcn_wmma_f32_16x16x32_bf16(false, aqx, false, bkx,
                                                            (short)0, ax[h], false, false);
            bf16x16 aqy = *(const bf16x16*)(qy + qoff + c * 32);
            bf16x16 bky = *(const bf16x16*)(ky + koff + c * 32);
            ay[h] = __builtin_amdgcn_wmma_f32_16x16x32_bf16(false, aqy, false, bky,
                                                            (short)0, ay[h], false, false);
        }
    }

    #pragma unroll
    for (int i = 0; i < 8; ++i) {
        float lx[HH], ly[HH];
        float mx = -1e30f, my = -1e30f;
        #pragma unroll
        for (int h = 0; h < HH; ++h) {
            const float dx = ax[h][i] * ATT_SCALE;
            const float dy = ay[h][i] * ATT_SCALE;
            lx[h] = dx + g1 * dy;
            ly[h] = dy + g2 * dx;
            mx = fmaxf(mx, lx[h]);
            my = fmaxf(my, ly[h]);
        }
        float sx = 0.f, sy = 0.f;
        #pragma unroll
        for (int h = 0; h < HH; ++h) {
            lx[h] = __expf(lx[h] - mx); sx += lx[h];
            ly[h] = __expf(ly[h] - my); sy += ly[h];
        }
        const float rx = 1.f / sx, ry = 1.f / sy;

        // store: one base per (i); per-head step is constant ATTH_STRIDE
        const int n = n0 + i + 8 * half;
        const size_t o0 = ((size_t)(b * HH) * NN + n) * NN + m0 + lm;
        __bf16* px = attn + o0;
        __bf16* py = attn + ATT_ELEMS + o0;
        #pragma unroll
        for (int h = 0; h < HH; ++h) {
            px[(size_t)h * ATTH_STRIDE] = (__bf16)(lx[h] * rx);
            py[(size_t)h * ATTH_STRIDE] = (__bf16)(ly[h] * ry);
        }
    }
}

// ---------------------------------------------------------------------------
// Kernel 3: attn @ V via TDM double-buffering.
// Block = one (n-tile, b, h, stream); 4 waves cover the four D/16 column
// tiles. The Tensor Data Mover streams 32x64 bf16 V chunks into LDS
// (tensor_load_to_lds, synced with s_wait_tensorcnt + barrier) while waves
// run WMMAs on the previous chunk. attn A-operand is a 32B contiguous load.
// ---------------------------------------------------------------------------
__global__ __launch_bounds__(128)
void av_kernel(const __bf16* __restrict__ qkv, const __bf16* __restrict__ attn,
               float* __restrict__ out)
{
    __shared__ __bf16 vtile[2][32][64];               // 2 x 4 KB double buffer

    const int lane = threadIdx.x;
    const int wave = threadIdx.y;                     // d-tile 0..3
    const int half = lane >> 4;
    const int lm   = lane & 15;
    const int n0 = blockIdx.x * 16;
    const int bh = blockIdx.y;                        // 0..47
    const int t  = blockIdx.z;                        // 0:x 1:y
    const int d0 = wave * 16;

    const __bf16* V  = qkv + ((size_t)t * 3 + 2) * QKV_ELEMS + (size_t)bh * NN * DD;
    const __bf16* Am = attn + (size_t)t * ATT_ELEMS;

    const unsigned long long gbase = (unsigned long long)(size_t)V;  // byte addr
    const unsigned int ldsoff[2] = {
        (unsigned int)(size_t)&vtile[0][0][0],
        (unsigned int)(size_t)&vtile[1][0][0]
    };

    // TDM: 2-D tensor, data_size=2B, tensor_dim0=64 (d), dim0_stride=64,
    // tensor_dim1=1024 (m), tile = 64 x 32, D# per ISA ch.8 bit layout.
    auto tdm_issue = [&](int kk, unsigned int lds_addr) {
        const unsigned long long ga = gbase + (unsigned long long)kk * DD * 2;
        u32x4 g0 = {};
        g0[0] = 1u;                                            // count=1 (valid, load)
        g0[1] = lds_addr;                                      // lds_addr [63:32]
        g0[2] = (unsigned int)(ga & 0xffffffffu);              // global_addr lo
        g0[3] = (unsigned int)((ga >> 32) & 0x1ffffffu)        // global_addr hi
              | (2u << 30);                                    // type=2 (image)
        i32x8 g1 = {};
        g1[0] = (int)(1u << 16);        // workgroup_mask=0, data_size=1 (2 bytes)
        g1[1] = (int)(64u << 16);       // tensor_dim0[15:0] in bits [63:48]
        g1[2] = (int)(1024u << 16);     // tensor_dim1[15:0] in bits [95:80]
        g1[3] = (int)(64u << 16);       // tile_dim0 = 64 in bits [127:112]
        g1[4] = 32;                     // tile_dim1 = 32 in bits [143:128]
        g1[5] = 64;                     // tensor_dim0_stride[31:0]
        g1[6] = 0;
        g1[7] = 0;
        i32x4 gz4 = {};
        i32x8 gz8 = {};
        __builtin_amdgcn_tensor_load_to_lds(g0, g1, gz4, gz4, gz8, 0);
    };

    if (wave == 0) tdm_issue(0, ldsoff[0]);

    f32x8 acc = {};
    const __bf16* arow = Am + ((size_t)bh * NN + n0 + lm) * NN + 16 * half;

    for (int it = 0; it < NN / 32; ++it) {
        const int kk  = it * 32;
        const int cur = it & 1;

        if (wave == 0) __builtin_amdgcn_s_wait_tensorcnt(0);  // buf[cur] ready
        __syncthreads();
        if (wave == 0 && kk + 32 < NN)                        // prefetch next chunk
            tdm_issue(kk + 32, ldsoff[1 - cur]);

        bf16x16 a = *(const bf16x16*)(arow + kk);             // 32B aligned
        bf16x16 bv;                                           // B operand from LDS
        #pragma unroll
        for (int e = 0; e < 16; ++e) bv[e] = vtile[cur][16 * half + e][d0 + lm];

        acc = __builtin_amdgcn_wmma_f32_16x16x32_bf16(false, a, false, bv,
                                                      (short)0, acc, false, false);
        if (kk + 32 < NN) __builtin_prefetch(arow + kk + 32, 0, 3);
    }

    // store: consecutive n rows -> constant OUTD stride from one base
    const int b = bh / HH, h = bh - b * HH;
    float* op = out + (size_t)t * BB * NN * OUTD
              + ((size_t)(b * NN + n0 + 8 * half)) * OUTD + h * DD + d0 + lm;
    #pragma unroll
    for (int i = 0; i < 8; ++i)
        op[(size_t)i * OUTD] = acc[i];
}

// ---------------------------------------------------------------------------
extern "C" void kernel_launch(void* const* d_in, const int* in_sizes, int n_in,
                              void* d_out, int out_size, void* d_ws, size_t ws_size,
                              hipStream_t stream)
{
    const float* x  = (const float*)d_in[0];
    const float* y  = (const float*)d_in[1];
    const float* Wx = (const float*)d_in[2];
    const float* bx = (const float*)d_in[3];
    const float* Wy = (const float*)d_in[4];
    const float* by = (const float*)d_in[5];
    const float* g1 = (const float*)d_in[6];
    const float* g2 = (const float*)d_in[7];
    float* out = (float*)d_out;

    // Workspace: [2][3] bf16 QKV head-major (37.7 MB) + [2] bf16 attn (201 MB)
    __bf16* qkv  = (__bf16*)d_ws;
    __bf16* attn = qkv + 6 * QKV_ELEMS;

    dim3 blk(32, 4);
    qkv_kernel<<<dim3(BB * NN / 16, OUTC / 64, 2), blk, 0, stream>>>(
        x, y, Wx, bx, Wy, by, qkv);
    scores_softmax_kernel<<<dim3(NN / 16, NN / 64, BB), blk, 0, stream>>>(
        qkv, g1, g2, attn);
    av_kernel<<<dim3(NN / 16, BB * HH, 2), blk, 0, stream>>>(qkv, attn, out);
}
